// MixtralSparTAMoeBlock_7164005450199
// MI455X (gfx1250) — compile-verified, matrix operand order checked by reading, not run on hardware
//
#include <hip/hip_runtime.h>

#define T_TOK 4096
#define H_DIM 2048
#define F_DIM 7168
#define NE    8
#define LDA   40   // LDS row stride in bf16 elements (32 data + 8 pad)

typedef unsigned short u16;
typedef unsigned int   u32;

typedef __attribute__((ext_vector_type(16))) __bf16 v16bf;
typedef __attribute__((ext_vector_type(8)))  float  v8f;
typedef __attribute__((ext_vector_type(4)))  int    v4i;

union Frag { uint4 q[2]; v16bf v; };

#if __has_builtin(__builtin_amdgcn_global_load_async_to_lds_b128)
#define HAVE_ASYNC_LDS 1
#endif

__device__ __forceinline__ u16 f2bf(float f) {
  u32 u = __float_as_uint(f);
  u32 r = u + 0x7FFFu + ((u >> 16) & 1u);   // round-to-nearest-even
  return (u16)(r >> 16);
}

// Pack two fp32 -> packed bf16x2 (low = first arg).
// v_cvt_pk_bf16_f32 confirmed on gfx1250; clang-22 lacks the builtin, so asm.
__device__ __forceinline__ u32 pack_bf2(float lo, float hi) {
#if __has_builtin(__builtin_amdgcn_cvt_pk_bf16_f32)
  typedef __attribute__((ext_vector_type(2))) __bf16 v2bf;
  union { v2bf v; u32 u; } cv;
  cv.v = __builtin_amdgcn_cvt_pk_bf16_f32(lo, hi);
  return cv.u;
#elif defined(__gfx1250__)
  u32 r;
  asm("v_cvt_pk_bf16_f32 %0, %1, %2" : "=v"(r) : "v"(lo), "v"(hi));
  return r;
#else
  return (u32)f2bf(lo) | ((u32)f2bf(hi) << 16);
#endif
}

// Async global->LDS 16B copy (ASYNCcnt-tracked) with synchronous fallback.
__device__ __forceinline__ void async_copy_b128(const void* g, void* l) {
#ifdef HAVE_ASYNC_LDS
  __builtin_amdgcn_global_load_async_to_lds_b128(
      (__attribute__((address_space(1))) v4i*)(void*)g,
      (__attribute__((address_space(3))) v4i*)l, 0, 0);
#else
  *(uint4*)l = *(const uint4*)g;
#endif
}

__device__ __forceinline__ void wait_async() {
#ifdef HAVE_ASYNC_LDS
#if __has_builtin(__builtin_amdgcn_s_wait_asynccnt)
  __builtin_amdgcn_s_wait_asynccnt(0);
#else
  asm volatile("s_wait_asynccnt 0" ::: "memory");
#endif
#endif
}

// Load one 16x32 bf16 operand tile (A or B^T) from LDS into a WMMA fragment.
// ISA layout: lanes 0-15 -> rows 0-15 with K = 0..7 (VGPR0-3) and 16..23 (VGPR4-7);
//             lanes 16-31 -> rows 0-15 with K = 8..15 and 24..31.
__device__ __forceinline__ void load_frag(const u16* base, int row0, int lane, Frag& f) {
  int r  = row0 + (lane & 15);
  int kb = (lane >> 4) << 3;                 // 0 or 8
  const u16* p = base + r * LDA + kb;
  f.q[0] = *reinterpret_cast<const uint4*>(p);        // K = kb .. kb+7
  f.q[1] = *reinterpret_cast<const uint4*>(p + 16);   // K = kb+16 .. kb+23
}

// ---------------------------------------------------------------- zero / init
__global__ void zero_out_kernel(float* __restrict__ out, int n4, int* __restrict__ counts) {
  int i = blockIdx.x * blockDim.x + threadIdx.x;
  if (i < NE) counts[i] = 0;
  float4 z = make_float4(0.f, 0.f, 0.f, 0.f);
  float4* o = (float4*)out;
  for (int j = i; j < n4; j += gridDim.x * blockDim.x) o[j] = z;
}

// ---------------------------------------------------------------- x -> bf16
__global__ void cvt_x_kernel(const float* __restrict__ x, u16* __restrict__ xb) {
  int i = blockIdx.x * blockDim.x + threadIdx.x;     // one float4 per thread
  float4 v = ((const float4*)x)[i];
  ((uint2*)xb)[i] = make_uint2(pack_bf2(v.x, v.y), pack_bf2(v.z, v.w));
}

// ---------------------------------------------------------------- router
__global__ void router_kernel(const float* __restrict__ x, const float* __restrict__ gw,
                              float* __restrict__ logits, float* __restrict__ tokw,
                              int* __restrict__ counts, int* __restrict__ etok) {
  int lane = threadIdx.x & 31;
  int wv   = threadIdx.x >> 5;
  int t    = blockIdx.x * 4 + wv;          // one wave per token
  float acc[NE];
#pragma unroll
  for (int e = 0; e < NE; ++e) acc[e] = 0.f;
  const float* xr = x + (size_t)t * H_DIM;
  for (int h = lane; h < H_DIM; h += 32) {
    float xv = xr[h];
#pragma unroll
    for (int e = 0; e < NE; ++e) acc[e] += xv * gw[e * H_DIM + h];
  }
#pragma unroll
  for (int e = 0; e < NE; ++e) {
#pragma unroll
    for (int off = 16; off >= 1; off >>= 1) acc[e] += __shfl_xor(acc[e], off, 32);
  }
  if (lane == 0) {
    float mx = acc[0];
#pragma unroll
    for (int e = 1; e < NE; ++e) mx = fmaxf(mx, acc[e]);
    float ex[NE];
#pragma unroll
    for (int e = 0; e < NE; ++e) ex[e] = __expf(acc[e] - mx);
    int i1 = 0; float v1 = ex[0];
#pragma unroll
    for (int e = 1; e < NE; ++e) if (ex[e] > v1) { v1 = ex[e]; i1 = e; }
    int i2 = -1; float v2 = -1.f;
#pragma unroll
    for (int e = 0; e < NE; ++e) if (e != i1 && ex[e] > v2) { v2 = ex[e]; i2 = e; }
    float inv = 1.f / (v1 + v2);           // softmax denom cancels in the ratio
    tokw[t * NE + i1] = v1 * inv;
    tokw[t * NE + i2] = v2 * inv;
    int p1 = atomicAdd(&counts[i1], 1); etok[i1 * T_TOK + p1] = t;
    int p2 = atomicAdd(&counts[i2], 1); etok[i2 * T_TOK + p2] = t;
#pragma unroll
    for (int e = 0; e < NE; ++e) logits[t * NE + e] = acc[e];
  }
}

// ---------------------------------------------------------------- scan
__global__ void scan_kernel(const int* __restrict__ counts, int* __restrict__ base) {
  if (threadIdx.x == 0) {
    int s = 0;
    for (int e = 0; e < NE; ++e) { base[e] = s; s += counts[e]; }
  }
}

// ---------------------------------------------------------------- FFN stage 1: h = silu(x w1^T) * (x w3^T)
// grid: (m-tiles fastest, n-tiles, expert) so concurrent m-tiles share weight
// strips through L2 -> HBM weight traffic stays near the 1x floor.
__global__ void __launch_bounds__(256)
moe_ffn1_kernel(const u16* __restrict__ xb, const float* __restrict__ w1,
                const float* __restrict__ w3, const int* __restrict__ etok,
                const int* __restrict__ counts, const int* __restrict__ ebase,
                u16* __restrict__ hbuf) {
  const int e = blockIdx.z;
  const int cnt = counts[e];
  const int tileM = blockIdx.x * 64;      // token rows (fastest dim)
  if (tileM >= cnt) return;
  const int tileN = blockIdx.y * 128;     // F columns
  const int hbase = ebase[e];

  __shared__ __align__(16) u16 As [2][64  * LDA];
  __shared__ __align__(16) u16 B1s[2][128 * LDA];
  __shared__ __align__(16) u16 B3s[2][128 * LDA];

  const int tid  = threadIdx.x;
  const int lane = tid & 31;
  const int wv   = tid >> 5;    // 8 waves: 2 (M) x 4 (N)
  const int m_w  = wv >> 2;
  const int n_w  = wv & 3;

  v8f acc1[2][2] = {};
  v8f acc3[2][2] = {};

  // A tile: 64 rows x 32 bf16 from gathered tokens (one b128 async copy per thread)
  const int ar = tid >> 2;
  const int ac = (tid & 3) << 3;
  const int arie = tileM + ar;
  const int atok = etok[e * T_TOK + (arie < cnt ? arie : 0)];
  const u16* asrc = xb + (size_t)atok * H_DIM + ac;
  // B tiles: 128 rows x 32 fp32 -> bf16 (16 floats per thread per matrix)
  const int br = tid >> 1;
  const int bc = (tid & 1) << 4;
  const float* b1src = w1 + ((size_t)e * F_DIM + tileN + br) * H_DIM + bc;
  const float* b3src = w3 + ((size_t)e * F_DIM + tileN + br) * H_DIM + bc;

  auto load_stage = [&](int st, int k0) {
    async_copy_b128(asrc + k0, &As[st][ar * LDA + ac]);
#pragma unroll
    for (int m = 0; m < 2; ++m) {
      const float* src = m ? b3src : b1src;
      u16* dst = m ? &B3s[st][br * LDA + bc] : &B1s[st][br * LDA + bc];
      float4 f0 = ((const float4*)(src + k0))[0];
      float4 f1 = ((const float4*)(src + k0))[1];
      float4 f2 = ((const float4*)(src + k0))[2];
      float4 f3 = ((const float4*)(src + k0))[3];
      uint4 p0, p1;
      p0.x = pack_bf2(f0.x, f0.y);
      p0.y = pack_bf2(f0.z, f0.w);
      p0.z = pack_bf2(f1.x, f1.y);
      p0.w = pack_bf2(f1.z, f1.w);
      p1.x = pack_bf2(f2.x, f2.y);
      p1.y = pack_bf2(f2.z, f2.w);
      p1.z = pack_bf2(f3.x, f3.y);
      p1.w = pack_bf2(f3.z, f3.w);
      ((uint4*)dst)[0] = p0;
      ((uint4*)dst)[1] = p1;
    }
  };

  load_stage(0, 0);
  wait_async();
  __syncthreads();

  const int NK = H_DIM / 32;
  for (int kk = 0; kk < NK; ++kk) {
    const int cur = kk & 1;
    if (kk + 1 < NK) load_stage(cur ^ 1, (kk + 1) * 32);
    Frag a[2], b1[2], b3[2];
#pragma unroll
    for (int fm = 0; fm < 2; ++fm) load_frag(As[cur], m_w * 32 + fm * 16, lane, a[fm]);
#pragma unroll
    for (int fn = 0; fn < 2; ++fn) {
      load_frag(B1s[cur], n_w * 32 + fn * 16, lane, b1[fn]);
      load_frag(B3s[cur], n_w * 32 + fn * 16, lane, b3[fn]);
    }
#pragma unroll
    for (int fm = 0; fm < 2; ++fm)
#pragma unroll
      for (int fn = 0; fn < 2; ++fn) {
        acc1[fm][fn] = __builtin_amdgcn_wmma_f32_16x16x32_bf16(
            false, a[fm].v, false, b1[fn].v, (short)0, acc1[fm][fn], false, false);
        acc3[fm][fn] = __builtin_amdgcn_wmma_f32_16x16x32_bf16(
            false, a[fm].v, false, b3[fn].v, (short)0, acc3[fm][fn], false, false);
      }
    wait_async();
    __syncthreads();
  }

  // SwiGLU epilogue, write bf16 h rows (compacted per expert)
#pragma unroll
  for (int fm = 0; fm < 2; ++fm)
#pragma unroll
    for (int fn = 0; fn < 2; ++fn)
#pragma unroll
      for (int j = 0; j < 8; ++j) {
        float a1 = acc1[fm][fn][j];
        float a3 = acc3[fm][fn][j];
        float s  = a1 / (1.f + __expf(-a1));
        float hv = s * a3;
        int ml = j + ((lane >> 4) << 3);
        int nl = lane & 15;
        int row = tileM + m_w * 32 + fm * 16 + ml;
        int col = tileN + n_w * 32 + fn * 16 + nl;
        if (row < cnt) hbuf[(size_t)(hbase + row) * F_DIM + col] = f2bf(hv);
      }
}

// ---------------------------------------------------------------- FFN stage 2: out += (h w2^T) * route_w
__global__ void __launch_bounds__(256)
moe_ffn2_kernel(const u16* __restrict__ hbuf, const float* __restrict__ w2,
                const int* __restrict__ etok, const int* __restrict__ counts,
                const int* __restrict__ ebase, const float* __restrict__ tokw,
                float* __restrict__ out) {
  const int e = blockIdx.z;
  const int cnt = counts[e];
  const int tileM = blockIdx.x * 64;      // h rows (fastest dim, shares w2 strips)
  if (tileM >= cnt) return;
  const int tileN = blockIdx.y * 128;     // H columns
  const int hbase = ebase[e];

  __shared__ __align__(16) u16 As[2][64  * LDA];
  __shared__ __align__(16) u16 Bs[2][128 * LDA];

  const int tid  = threadIdx.x;
  const int lane = tid & 31;
  const int wv   = tid >> 5;
  const int m_w  = wv >> 2;
  const int n_w  = wv & 3;

  v8f acc[2][2] = {};

  const int ar = tid >> 2;
  const int ac = (tid & 3) << 3;
  const int arie = tileM + ar;
  const u16* asrc = hbuf + (size_t)(hbase + (arie < cnt ? arie : 0)) * F_DIM + ac;
  const int br = tid >> 1;
  const int bc = (tid & 1) << 4;
  const float* bsrc = w2 + ((size_t)e * H_DIM + tileN + br) * F_DIM + bc;

  auto load_stage = [&](int st, int k0) {
    async_copy_b128(asrc + k0, &As[st][ar * LDA + ac]);
    float4 f0 = ((const float4*)(bsrc + k0))[0];
    float4 f1 = ((const float4*)(bsrc + k0))[1];
    float4 f2 = ((const float4*)(bsrc + k0))[2];
    float4 f3 = ((const float4*)(bsrc + k0))[3];
    uint4 p0, p1;
    p0.x = pack_bf2(f0.x, f0.y);
    p0.y = pack_bf2(f0.z, f0.w);
    p0.z = pack_bf2(f1.x, f1.y);
    p0.w = pack_bf2(f1.z, f1.w);
    p1.x = pack_bf2(f2.x, f2.y);
    p1.y = pack_bf2(f2.z, f2.w);
    p1.z = pack_bf2(f3.x, f3.y);
    p1.w = pack_bf2(f3.z, f3.w);
    ((uint4*)&Bs[st][br * LDA + bc])[0] = p0;
    ((uint4*)&Bs[st][br * LDA + bc])[1] = p1;
  };

  load_stage(0, 0);
  wait_async();
  __syncthreads();

  const int NK = F_DIM / 32;              // 224
  for (int kk = 0; kk < NK; ++kk) {
    const int cur = kk & 1;
    if (kk + 1 < NK) load_stage(cur ^ 1, (kk + 1) * 32);
    Frag a[2], b[2];
#pragma unroll
    for (int fm = 0; fm < 2; ++fm) load_frag(As[cur], m_w * 32 + fm * 16, lane, a[fm]);
#pragma unroll
    for (int fn = 0; fn < 2; ++fn) load_frag(Bs[cur], n_w * 32 + fn * 16, lane, b[fn]);
#pragma unroll
    for (int fm = 0; fm < 2; ++fm)
#pragma unroll
      for (int fn = 0; fn < 2; ++fn)
        acc[fm][fn] = __builtin_amdgcn_wmma_f32_16x16x32_bf16(
            false, a[fm].v, false, b[fn].v, (short)0, acc[fm][fn], false, false);
    wait_async();
    __syncthreads();
  }

  // Scale by routing weight and scatter-accumulate (token appears in 2 experts)
#pragma unroll
  for (int fm = 0; fm < 2; ++fm)
#pragma unroll
    for (int fn = 0; fn < 2; ++fn)
#pragma unroll
      for (int j = 0; j < 8; ++j) {
        int ml = j + ((lane >> 4) << 3);
        int nl = lane & 15;
        int rie = tileM + m_w * 32 + fm * 16 + ml;
        if (rie < cnt) {
          int tok   = etok[e * T_TOK + rie];
          float wgt = tokw[tok * NE + e];
          int col   = tileN + n_w * 32 + fn * 16 + nl;
          atomicAdd(&out[(size_t)tok * H_DIM + col], acc[fm][fn][j] * wgt);
        }
      }
}

// ---------------------------------------------------------------- launch
extern "C" void kernel_launch(void* const* d_in, const int* in_sizes, int n_in,
                              void* d_out, int out_size, void* d_ws, size_t ws_size,
                              hipStream_t stream) {
  const float* x      = (const float*)d_in[0];   // [T, H]
  const float* gate_w = (const float*)d_in[1];   // [E, H]
  const float* w1     = (const float*)d_in[2];   // [E, F, H]
  const float* w2     = (const float*)d_in[3];   // [E, H, F]
  const float* w3     = (const float*)d_in[4];   // [E, F, H]
  float* out = (float*)d_out;                              // out [T*H] ++ logits [T*E]
  float* logits = out + (size_t)T_TOK * H_DIM;

  // workspace carve (all 16B aligned)
  char* ws = (char*)d_ws;
  int*   counts = (int*)ws;                                           //  8 ints
  int*   ebase  = (int*)(ws + 32);                                    //  8 ints
  int*   etok   = (int*)(ws + 64);                                    //  E*T ints  (128 KB)
  float* tokw   = (float*)(ws + 64 + (size_t)NE * T_TOK * 4);         //  T*E floats (128 KB)
  u16*   xb     = (u16*)(ws + 64 + (size_t)NE * T_TOK * 4
                             + (size_t)T_TOK * NE * 4);               //  T*H bf16 (16 MB)
  u16*   hbuf   = xb + (size_t)T_TOK * H_DIM;                         //  2T*F bf16 (117 MB)

  const int n4 = T_TOK * H_DIM / 4;
  zero_out_kernel<<<4096, 256, 0, stream>>>(out, n4, counts);
  cvt_x_kernel<<<n4 / 256, 256, 0, stream>>>(x, xb);
  router_kernel<<<T_TOK / 4, 128, 0, stream>>>(x, gate_w, logits, tokw, counts, etok);
  scan_kernel<<<1, 32, 0, stream>>>(counts, ebase);

  dim3 g1(T_TOK / 64, F_DIM / 128, NE);   // m-tiles fastest: 64 x 56 x 8
  moe_ffn1_kernel<<<g1, 256, 0, stream>>>(xb, w1, w3, etok, counts, ebase, hbuf);

  dim3 g2(T_TOK / 64, H_DIM / 128, NE);   // m-tiles fastest: 64 x 16 x 8
  moe_ffn2_kernel<<<g2, 256, 0, stream>>>(hbuf, w2, etok, counts, ebase, tokw, out);
}
